// GNN_4861902979261
// MI455X (gfx1250) — matrix operand order repaired and checked by reading, not compiled, not run on hardware
//
#include <hip/hip_runtime.h>
#include <hip/hip_bf16.h>

// ---------------- constants (match reference shapes) ----------------
#define FIN 128
#define H1  64
#define H2  16
#define NC  10
#define NEG_SLOPE 0.2f

typedef __attribute__((ext_vector_type(2)))  float    v2f;
typedef __attribute__((ext_vector_type(8)))  float    v8f;
typedef __attribute__((ext_vector_type(16))) _Float16 v16h;

// ---------------- order-preserving float<->uint encoding for atomicMax ----
__device__ __forceinline__ unsigned fenc(float f) {
  unsigned u = __float_as_uint(f);
  return (u & 0x80000000u) ? ~u : (u | 0x80000000u);
}
__device__ __forceinline__ float fdec(unsigned e) {
  return __uint_as_float((e & 0x80000000u) ? (e & 0x7fffffffu) : ~e);
}

__device__ __forceinline__ void edge_nodes(const long long* __restrict__ ei,
                                           long long e, long long E,
                                           int& src, int& dst) {
  if (e < E) { src = (int)ei[e]; dst = (int)ei[E + e]; }
  else       { src = dst = (int)(e - E); }   // self loops appended
}

// ---------------- WMMA GEMM: D[nrows,FOUT] = A[nrows,KDIM] @ B[KDIM,FOUT] ----
// One wave per 16x16 output tile. grid.x = row tiles, grid.y = col tiles.
#if __has_builtin(__builtin_amdgcn_wmma_f32_16x16x4_f32)

template <int KDIM, int FOUT>
__global__ void gemm_wmma(const float* __restrict__ A, const float* __restrict__ B,
                          float* __restrict__ D, int nrows) {
  const int rowBase = blockIdx.x * 16;
  const int colBase = blockIdx.y * 16;
  if (rowBase >= nrows) return;
  const int lane = threadIdx.x & 31;
  const int mn   = lane & 15;          // M for A, N for B/C/D
  const int kh   = (lane >> 4) * 2;    // K pair selected by lane half
  v8f acc = {};
  const float* arow = A + (size_t)(rowBase + mn) * KDIM;
  for (int k0 = 0; k0 < KDIM; k0 += 4) {
    v2f a, b;
    a.x = arow[k0 + kh];
    a.y = arow[k0 + kh + 1];
    b.x = B[(size_t)(k0 + kh)     * FOUT + colBase + mn];
    b.y = B[(size_t)(k0 + kh + 1) * FOUT + colBase + mn];
    acc = __builtin_amdgcn_wmma_f32_16x16x4_f32(false, a, false, b,
                                                (short)0, acc, false, false);
  }
  const int mAdd = (lane >> 4) * 8;
#pragma unroll
  for (int v = 0; v < 8; ++v)
    D[(size_t)(rowBase + v + mAdd) * FOUT + colBase + mn] = acc[v];
}

#else  // fallback: codegen-confirmed f16 WMMA (convert fp32 -> fp16 on the fly)

template <int KDIM, int FOUT>
__global__ void gemm_wmma(const float* __restrict__ A, const float* __restrict__ B,
                          float* __restrict__ D, int nrows) {
  const int rowBase = blockIdx.x * 16;
  const int colBase = blockIdx.y * 16;
  if (rowBase >= nrows) return;
  const int lane = threadIdx.x & 31;
  const int mn = lane & 15;
  const int hi = lane >> 4;
  v8f acc = {};
  const float* arow = A + (size_t)(rowBase + mn) * KDIM;
  for (int k0 = 0; k0 < KDIM; k0 += 32) {
    v16h av, bv;
#pragma unroll
    for (int j = 0; j < 16; ++j) {
      int ka = k0 + hi * 16 + j;                          // contiguous k = 16*hi + j
      av[j] = (_Float16)arow[ka];
      int kb = k0 + hi * 16 + j;                          // same contiguous k order as A
      bv[j] = (_Float16)B[(size_t)kb * FOUT + colBase + mn];
    }
    acc = __builtin_amdgcn_wmma_f32_16x16x32_f16(false, av, false, bv,
                                                 (short)0, acc, false, false);
  }
  const int mAdd = hi * 8;
#pragma unroll
  for (int v = 0; v < 8; ++v)
    D[(size_t)(rowBase + v + mAdd) * FOUT + colBase + mn] = acc[v];
}

#endif

// ---------------- per-node attention dot products (wave per node) ----------
template <int F>
__global__ void dots_kernel(const float* __restrict__ h,
                            const float* __restrict__ a_src,
                            const float* __restrict__ a_dst,
                            float* __restrict__ ss, float* __restrict__ sd, int N) {
  const int node = blockIdx.x * (blockDim.x >> 5) + (threadIdx.x >> 5);
  if (node >= N) return;
  const int lane = threadIdx.x & 31;
  const float* row = h + (size_t)node * F;
  float vs = 0.f, vd = 0.f;
  for (int f = lane; f < F; f += 32) {
    float hv = row[f];
    vs += hv * a_src[f];
    vd += hv * a_dst[f];
  }
#pragma unroll
  for (int off = 16; off; off >>= 1) {
    vs += __shfl_down(vs, off, 32);
    vd += __shfl_down(vd, off, 32);
  }
  if (lane == 0) { ss[node] = vs; sd[node] = vd; }
}

// ---------------- edge pass 1: segment max (L2 atomics on encoded floats) ---
__global__ void edge_max_kernel(const long long* __restrict__ ei,
                                const float* __restrict__ ss, const float* __restrict__ sd,
                                unsigned* __restrict__ menc,
                                long long E, long long Etot) {
  long long e = (long long)blockIdx.x * blockDim.x + threadIdx.x;
  if (e >= Etot) return;
  int src, dst; edge_nodes(ei, e, E, src, dst);
  float s = ss[src] + sd[dst];
  s = s > 0.f ? s : NEG_SLOPE * s;
  atomicMax(&menc[dst], fenc(s));
}

// ---------------- edge pass 2: segment sum of exp(e - m[dst]) --------------
__global__ void edge_sum_kernel(const long long* __restrict__ ei,
                                const float* __restrict__ ss, const float* __restrict__ sd,
                                const unsigned* __restrict__ menc, float* __restrict__ z,
                                long long E, long long Etot) {
  long long e = (long long)blockIdx.x * blockDim.x + threadIdx.x;
  if (e >= Etot) return;
  int src, dst; edge_nodes(ei, e, E, src, dst);
  float s = ss[src] + sd[dst];
  s = s > 0.f ? s : NEG_SLOPE * s;
  atomicAdd(&z[dst], __expf(s - fdec(menc[dst])));
}

// ---------------- edge pass 3: weighted scatter-add of features ------------
// LPE lanes cooperate on one edge: coalesced gather of h[src,*], L2 scatter.
template <int F, int LPE>
__global__ void edge_agg_kernel(const long long* __restrict__ ei,
                                const float* __restrict__ ss, const float* __restrict__ sd,
                                const unsigned* __restrict__ menc, const float* __restrict__ z,
                                const float* __restrict__ h, float* __restrict__ acc,
                                long long E, long long Etot) {
  long long gid = (long long)blockIdx.x * blockDim.x + threadIdx.x;
  long long e   = gid / LPE;
  int sub       = (int)(gid % LPE);
  if (e >= Etot) return;
  int src, dst; edge_nodes(ei, e, E, src, dst);
  float s = ss[src] + sd[dst];
  s = s > 0.f ? s : NEG_SLOPE * s;
  float alpha = __expf(s - fdec(menc[dst])) / z[dst];
  const float* hrow = h + (size_t)src * F;
  float* arow = acc + (size_t)dst * F;
#pragma unroll
  for (int f = sub; f < F; f += LPE)
    atomicAdd(&arow[f], hrow[f] * alpha);
}

// ---------------- bias + ReLU ----------------------------------------------
template <int F>
__global__ void bias_relu_kernel(float* __restrict__ acc, const float* __restrict__ b,
                                 long long total) {
  long long i = (long long)blockIdx.x * blockDim.x + threadIdx.x;
  if (i >= total) return;
  float v = acc[i] + b[(int)(i & (F - 1))];
  acc[i] = v > 0.f ? v : 0.f;
}

// ---------------- classifier: logits + log_softmax (fused) -----------------
__global__ void classifier_kernel(const float* __restrict__ acc2,
                                  const float* __restrict__ b2,
                                  const float* __restrict__ Wc,
                                  const float* __restrict__ bc,
                                  float* __restrict__ out, int N) {
  int n = blockIdx.x * blockDim.x + threadIdx.x;
  if (n >= N) return;
  float hf[H2];
#pragma unroll
  for (int k = 0; k < H2; ++k) hf[k] = acc2[(size_t)n * H2 + k] + b2[k];
  float logit[NC];
  float mx = -3.402823466e38f;
#pragma unroll
  for (int c = 0; c < NC; ++c) {
    float v = bc[c];
#pragma unroll
    for (int k = 0; k < H2; ++k) v += hf[k] * Wc[k * NC + c];
    logit[c] = v;
    mx = v > mx ? v : mx;
  }
  float se = 0.f;
#pragma unroll
  for (int c = 0; c < NC; ++c) se += __expf(logit[c] - mx);
  float lse = mx + __logf(se);
#pragma unroll
  for (int c = 0; c < NC; ++c) out[(size_t)n * NC + c] = logit[c] - lse;
}

// ---------------- host-side orchestration ----------------------------------
extern "C" void kernel_launch(void* const* d_in, const int* in_sizes, int n_in,
                              void* d_out, int out_size, void* d_ws, size_t ws_size,
                              hipStream_t stream) {
  const float*     x   = (const float*)d_in[0];
  const long long* ei  = (const long long*)d_in[1];   // int64 [2, E]
  const float*     W1  = (const float*)d_in[2];
  const float*     a1s = (const float*)d_in[3];
  const float*     a1d = (const float*)d_in[4];
  const float*     b1  = (const float*)d_in[5];
  const float*     W2  = (const float*)d_in[6];
  const float*     a2s = (const float*)d_in[7];
  const float*     a2d = (const float*)d_in[8];
  const float*     b2  = (const float*)d_in[9];
  const float*     Wc  = (const float*)d_in[10];
  const float*     bc  = (const float*)d_in[11];
  float* out = (float*)d_out;

  const int       N    = in_sizes[0] / FIN;
  const long long E    = (long long)in_sizes[1] / 2;
  const long long Etot = E + N;

  // workspace carve-out (floats)
  float* ws = (float*)d_ws;
  size_t off = 0;
  float* h1  = ws + off; off += (size_t)N * H1;
  float* h2  = ws + off; off += (size_t)N * H2;
  float* s1s = ws + off; off += N;
  float* s1d = ws + off; off += N;
  float* s2s = ws + off; off += N;
  float* s2d = ws + off; off += N;
  float* zbase = ws + off;                 // everything below must start at 0
  float* acc1 = ws + off; off += (size_t)N * H1;
  float* acc2 = ws + off; off += (size_t)N * H2;
  float* z1   = ws + off; off += N;
  float* z2   = ws + off; off += N;
  unsigned* m1 = (unsigned*)(ws + off); off += N;
  unsigned* m2 = (unsigned*)(ws + off); off += N;
  size_t zeroBytes = (size_t)((ws + off) - zbase) * sizeof(float);
  hipMemsetAsync(zbase, 0, zeroBytes, stream);   // m_enc=0 decodes below any real score

  const int TB = 256;

  // ---- Layer 1 ----
  gemm_wmma<FIN, H1><<<dim3((N + 15) / 16, H1 / 16), 32, 0, stream>>>(x, W1, h1, N);
  dots_kernel<H1><<<(N + 7) / 8, TB, 0, stream>>>(h1, a1s, a1d, s1s, s1d, N);
  edge_max_kernel<<<(unsigned)((Etot + TB - 1) / TB), TB, 0, stream>>>(ei, s1s, s1d, m1, E, Etot);
  edge_sum_kernel<<<(unsigned)((Etot + TB - 1) / TB), TB, 0, stream>>>(ei, s1s, s1d, m1, z1, E, Etot);
  edge_agg_kernel<H1, 32><<<(unsigned)((Etot * 32 + TB - 1) / TB), TB, 0, stream>>>(
      ei, s1s, s1d, m1, z1, h1, acc1, E, Etot);
  bias_relu_kernel<H1><<<(unsigned)(((long long)N * H1 + TB - 1) / TB), TB, 0, stream>>>(
      acc1, b1, (long long)N * H1);

  // ---- Layer 2 ----
  gemm_wmma<H1, H2><<<dim3((N + 15) / 16, H2 / 16), 32, 0, stream>>>(acc1, W2, h2, N);
  dots_kernel<H2><<<(N + 7) / 8, TB, 0, stream>>>(h2, a2s, a2d, s2s, s2d, N);
  edge_max_kernel<<<(unsigned)((Etot + TB - 1) / TB), TB, 0, stream>>>(ei, s2s, s2d, m2, E, Etot);
  edge_sum_kernel<<<(unsigned)((Etot + TB - 1) / TB), TB, 0, stream>>>(ei, s2s, s2d, m2, z2, E, Etot);
  edge_agg_kernel<H2, 16><<<(unsigned)((Etot * 16 + TB - 1) / TB), TB, 0, stream>>>(
      ei, s2s, s2d, m2, z2, h2, acc2, E, Etot);

  // ---- Classifier + log_softmax (adds b2; no ReLU after layer 2) ----
  classifier_kernel<<<(N + TB - 1) / TB, TB, 0, stream>>>(acc2, b2, Wc, bc, out, N);
}
